// BinaryXnorExceptOutliersLinear_6511170421195
// MI455X (gfx1250) — matrix operand order repaired and checked by reading, not compiled, not run on hardware
//
#include <hip/hip_runtime.h>
#include <hip/hip_bf16.h>

typedef unsigned short u16;
typedef unsigned int u32;
typedef __attribute__((ext_vector_type(16))) __bf16 v16bf;
typedef __attribute__((ext_vector_type(8)))  float  v8f;
typedef __attribute__((ext_vector_type(4)))  int    v4i;

#define AS1 __attribute__((address_space(1)))
#define AS3 __attribute__((address_space(3)))

#if __has_builtin(__builtin_amdgcn_global_load_async_to_lds_b128)
#define HAVE_ASYNC_LDS 1
#else
#define HAVE_ASYNC_LDS 0
#endif

__device__ __forceinline__ void wait_async0() {
#if __has_builtin(__builtin_amdgcn_s_wait_asynccnt)
    __builtin_amdgcn_s_wait_asynccnt(0);
#else
    asm volatile("s_wait_asynccnt 0x0" ::: "memory");
#endif
}

// ---------------- bf16 convert (round-to-nearest-even) ----------------
__device__ __forceinline__ u16 f2bf(float f) {
    u32 u = __float_as_uint(f);
    u32 r = u + 0x7FFFu + ((u >> 16) & 1u);
    return (u16)(r >> 16);
}

// ---------------- workspace layout ----------------
// [0..63]   : 4 doubles  (sum, sumsq, keepsum, keepcnt)
// [64..255] : 4 floats   (lo, hi, scale, pad)
// [256.. ]  : wq bf16 [D_OUT*D_IN], then xq bf16 [M*D_IN]
#define WS_DSCAL(ws)  ((double*)(ws))
#define WS_FSCAL(ws)  ((float*)((char*)(ws) + 64))
#define WS_WQ(ws)     ((u16*)((char*)(ws) + 256))

// ---------------- stage 0: zero scalar accumulators ----------------
__global__ void k_init(double* dscal) {
    if (threadIdx.x < 4) dscal[threadIdx.x] = 0.0;
}

// ---------------- stage 1: sum & sumsq over weight ----------------
__global__ __launch_bounds__(256) void k_stats1(const float* __restrict__ w,
                                                unsigned n, double* dscal) {
    __shared__ double sh0[256];
    __shared__ double sh1[256];
    double s = 0.0, q = 0.0;
    for (unsigned i = blockIdx.x * 256u + threadIdx.x; i < n; i += gridDim.x * 256u) {
        float v = w[i];
        s += (double)v;
        q += (double)v * (double)v;
    }
    sh0[threadIdx.x] = s; sh1[threadIdx.x] = q;
    __syncthreads();
    for (int off = 128; off > 0; off >>= 1) {
        if ((int)threadIdx.x < off) {
            sh0[threadIdx.x] += sh0[threadIdx.x + off];
            sh1[threadIdx.x] += sh1[threadIdx.x + off];
        }
        __syncthreads();
    }
    if (threadIdx.x == 0) {
        atomicAdd(&dscal[0], sh0[0]);
        atomicAdd(&dscal[1], sh1[0]);
    }
}

__global__ void k_finalize1(const double* dscal, float* fscal, unsigned n) {
    double sum = dscal[0], sumsq = dscal[1];
    double mean = sum / (double)n;
    double var  = (sumsq - sum * sum / (double)n) / (double)(n - 1); // ddof=1
    double sd   = sqrt(var > 0.0 ? var : 0.0);
    fscal[0] = (float)(mean - 1.6 * sd);
    fscal[1] = (float)(mean + 1.6 * sd);
}

// ---------------- stage 2: binary scale over non-outliers ----------------
__global__ __launch_bounds__(256) void k_stats2(const float* __restrict__ w,
                                                unsigned n, double* dscal,
                                                const float* fscal) {
    __shared__ double sh0[256];
    __shared__ double sh1[256];
    const float lo = fscal[0], hi = fscal[1];
    double s = 0.0, c = 0.0;
    for (unsigned i = blockIdx.x * 256u + threadIdx.x; i < n; i += gridDim.x * 256u) {
        float v = w[i];
        bool keep = !((v < lo) || (v > hi));
        if (keep) { s += (double)fabsf(v); c += 1.0; }
    }
    sh0[threadIdx.x] = s; sh1[threadIdx.x] = c;
    __syncthreads();
    for (int off = 128; off > 0; off >>= 1) {
        if ((int)threadIdx.x < off) {
            sh0[threadIdx.x] += sh0[threadIdx.x + off];
            sh1[threadIdx.x] += sh1[threadIdx.x + off];
        }
        __syncthreads();
    }
    if (threadIdx.x == 0) {
        atomicAdd(&dscal[2], sh0[0]);
        atomicAdd(&dscal[3], sh1[0]);
    }
}

__global__ void k_finalize2(const double* dscal, float* fscal) {
    double c = dscal[3];
    fscal[2] = (float)(c > 0.0 ? dscal[2] / c : 0.0);
}

// ---------------- stage 3: quantize weight -> w_sim bf16 ----------------
__global__ __launch_bounds__(256) void k_quant_w(const float* __restrict__ w,
                                                 u16* __restrict__ wq,
                                                 unsigned n4, const float* fscal) {
    const float lo = fscal[0], hi = fscal[1], sc = fscal[2];
    unsigned i = blockIdx.x * 256u + threadIdx.x;
    if (i >= n4) return;
    float4 v = ((const float4*)w)[i];
    float e[4] = {v.x, v.y, v.z, v.w};
    u16 r[4];
#pragma unroll
    for (int j = 0; j < 4; ++j) {
        float x = e[j];
        bool outl = (x < lo) || (x > hi);
        float sgn = (float)((x > 0.0f) - (x < 0.0f)); // jnp.sign (0 at 0)
        float val = outl ? x : sgn * sc;
        r[j] = f2bf(val);
    }
    uint2 o;
    o.x = (u32)r[0] | ((u32)r[1] << 16);
    o.y = (u32)r[2] | ((u32)r[3] << 16);
    ((uint2*)wq)[i] = o;
}

// ---------------- stage 4: quantize activations -> bf16 ----------------
__global__ __launch_bounds__(256) void k_quant_x(const float* __restrict__ x,
                                                 u16* __restrict__ xq, unsigned n4) {
    unsigned i = blockIdx.x * 256u + threadIdx.x;
    if (i >= n4) return;
    float4 v = ((const float4*)x)[i];
    uint2 o;
    o.x = (u32)f2bf(v.x) | ((u32)f2bf(v.y) << 16);
    o.y = (u32)f2bf(v.z) | ((u32)f2bf(v.w) << 16);
    ((uint2*)xq)[i] = o;
}

// ---------------- stage 5: WMMA bf16 GEMM, async-LDS double-buffered ----------------
// out[M,N] = Xq[M,K] * Wq[N,K]^T + bias[N]
#define TILE_M 128
#define TILE_N 256
#define TILE_K 32
#define LDSS   40   // LDS row stride in bf16 elems: 32 data + 8 pad (80B, 16B aligned)

union FragU { uint4 u[2]; v16bf v; };

__device__ __forceinline__ v16bf load_a_frag(const u16* lds, int row, int khalf) {
    // ISA 16-bit A 16x32 layout: lanes<16 hold K{0..7,16..23}, lanes>=16 K{8..15,24..31}
    FragU f;
    f.u[0] = *(const uint4*)(lds + row * LDSS + khalf * 8);
    f.u[1] = *(const uint4*)(lds + row * LDSS + 16 + khalf * 8);
    return f.v;
}

__device__ __forceinline__ v16bf load_b_frag(const u16* lds, int row, int khalf) {
    // B 32x16: lanes<16 hold K{0..7,16..23}, lanes>=16 hold K{8..15,24..31} (column-major gather)
    FragU f;
    f.u[0] = *(const uint4*)(lds + row * LDSS + khalf * 8);
    f.u[1] = *(const uint4*)(lds + row * LDSS + 16 + khalf * 8);
    return f.v;
}

// Stage a ROWSx32 bf16 tile (ROWS*4 uint4s) from global into an LDS buffer.
template <int ROWS>
__device__ __forceinline__ void stage_tile(const u16* __restrict__ G, u16* S,
                                           int base_row, int kcol, int K, int t) {
#pragma unroll
    for (int i = 0; i < (ROWS * 4) / 256; ++i) {
        int idx = t + i * 256;
        int r = idx >> 2;
        int c = (idx & 3) * 8;
        const u16* gp = G + (size_t)(base_row + r) * K + kcol + c;
        u16* sp = S + r * LDSS + c;
#if HAVE_ASYNC_LDS
        __builtin_amdgcn_global_load_async_to_lds_b128(
            (AS1 v4i*)gp, (AS3 v4i*)sp, /*offset=*/0, /*cpol=*/0);
#else
        *(uint4*)sp = *(const uint4*)gp;
#endif
    }
}

__global__ __launch_bounds__(256) void k_gemm_wmma_bf16(
    const u16* __restrict__ A,   // [M,K] bf16 (activations)
    const u16* __restrict__ B,   // [N,K] bf16 (binarized weights, row = out-feature)
    const float* __restrict__ bias,
    float* __restrict__ out,
    int M, int N, int K) {

    __shared__ __attribute__((aligned(16))) u16 sA[2][TILE_M * LDSS]; // 2 x 10 KB
    __shared__ __attribute__((aligned(16))) u16 sB[2][TILE_N * LDSS]; // 2 x 20 KB

    const int t    = threadIdx.x;
    const int lane = t & 31;
    const int wave = t >> 5;
    const int m0 = blockIdx.y * TILE_M;
    const int n0 = blockIdx.x * TILE_N;
    const int wm = (wave & 1) * 64;   // 2 waves along M, 64 rows each
    const int wn = (wave >> 1) * 64;  // 4 waves along N, 64 cols each

    v8f acc[4][4];
#pragma unroll
    for (int a = 0; a < 4; ++a)
#pragma unroll
        for (int b = 0; b < 4; ++b)
            acc[a][b] = (v8f){0.f, 0.f, 0.f, 0.f, 0.f, 0.f, 0.f, 0.f};

    const int khalf = lane >> 4;
    const int rl    = lane & 15;

    // prologue: fill buffer 0
    stage_tile<TILE_M>(A, sA[0], m0, 0, K, t);
    stage_tile<TILE_N>(B, sB[0], n0, 0, K, t);
    wait_async0();
    __syncthreads();

    int buf = 0;
    for (int kt = 0; kt < K; kt += TILE_K, buf ^= 1) {
        const int nkt = kt + TILE_K;
        const bool more = nkt < K;
        // kick off DMA of the next K-tile into the other buffer
        if (more) {
            stage_tile<TILE_M>(A, sA[buf ^ 1], m0, nkt, K, t);
            stage_tile<TILE_N>(B, sB[buf ^ 1], n0, nkt, K, t);
        }

        // compute 64x64 per wave from current buffer: 16 x v_wmma
        v16bf af[4], bfr[4];
#pragma unroll
        for (int tm = 0; tm < 4; ++tm) af[tm]  = load_a_frag(sA[buf], wm + tm * 16 + rl, khalf);
#pragma unroll
        for (int tn = 0; tn < 4; ++tn) bfr[tn] = load_b_frag(sB[buf], wn + tn * 16 + rl, khalf);

#pragma unroll
        for (int tm = 0; tm < 4; ++tm)
#pragma unroll
            for (int tn = 0; tn < 4; ++tn)
                acc[tm][tn] = __builtin_amdgcn_wmma_f32_16x16x32_bf16(
                    /*neg_a=*/false, af[tm], /*neg_b=*/false, bfr[tn],
                    /*c_mod=*/(short)0, acc[tm][tn],
                    /*reuse_a=*/false, /*reuse_b=*/false);

        if (more) wait_async0();  // our DMA into buf^1 has landed in LDS
        __syncthreads();          // everyone done reading buf / writing buf^1
    }

    // epilogue: C/D layout -> lane%16 = column, (lane/16)*8 + vgpr = row; fuse bias
#pragma unroll
    for (int tn = 0; tn < 4; ++tn) {
        int col = n0 + wn + tn * 16 + rl;
        float bv = bias[col];
#pragma unroll
        for (int tm = 0; tm < 4; ++tm) {
            int mb = m0 + wm + tm * 16 + khalf * 8;
#pragma unroll
            for (int r = 0; r < 8; ++r) {
                out[(size_t)(mb + r) * N + col] = acc[tm][tn][r] + bv;
            }
        }
    }
}

// ---------------- launch ----------------
extern "C" void kernel_launch(void* const* d_in, const int* in_sizes, int n_in,
                              void* d_out, int out_size, void* d_ws, size_t ws_size,
                              hipStream_t stream) {
    const float* x    = (const float*)d_in[0];   // [B,S,D_IN] f32
    const float* w    = (const float*)d_in[1];   // [D_OUT,D_IN] f32
    const float* bias = (const float*)d_in[2];   // [D_OUT] f32
    float* out = (float*)d_out;

    const int D_OUT = in_sizes[2];               // 4096
    const int D_IN  = in_sizes[1] / D_OUT;       // 4096
    const int Mrows = in_sizes[0] / D_IN;        // 8192 (= B*S)
    const unsigned nW = (unsigned)in_sizes[1];
    const unsigned nX = (unsigned)in_sizes[0];

    double* dscal = WS_DSCAL(d_ws);
    float*  fscal = WS_FSCAL(d_ws);
    u16*    wq    = WS_WQ(d_ws);
    u16*    xq    = wq + (size_t)D_OUT * D_IN;

    k_init<<<1, 64, 0, stream>>>(dscal);
    k_stats1<<<2048, 256, 0, stream>>>(w, nW, dscal);
    k_finalize1<<<1, 1, 0, stream>>>(dscal, fscal, nW);
    k_stats2<<<2048, 256, 0, stream>>>(w, nW, dscal, fscal);
    k_finalize2<<<1, 1, 0, stream>>>(dscal, fscal);

    k_quant_w<<<(nW / 4 + 255) / 256, 256, 0, stream>>>(w, wq, nW / 4, fscal);
    k_quant_x<<<(nX / 4 + 255) / 256, 256, 0, stream>>>(x, xq, nX / 4);

    dim3 grid(D_OUT / TILE_N, Mrows / TILE_M);
    k_gemm_wmma_bf16<<<grid, 256, 0, stream>>>(xq, wq, bias, out, Mrows, D_OUT, D_IN);
}
